// Module1_F_9028021256215
// MI455X (gfx1250) — matrix operand order repaired and checked by reading, not compiled, not run on hardware
//
#include <hip/hip_runtime.h>
#include <hip/hip_bf16.h>

// ---------------------------------------------------------------------------
// Shapes (compile-time constants from the reference)
// ---------------------------------------------------------------------------
#define BATCH 2
#define DIM   256
#define HEADS 8
#define HH    128
#define WW    128
#define HW    (HH*WW)              // 16384
#define NE    ((size_t)BATCH*DIM*HW)      // 8,388,608 elems per tensor
#define FW    65                   // rfft2 last-dim size
#define FPX   (HH*FW)              // 8320 freq pixels per channel
#define FE    ((size_t)BATCH*DIM*FPX)     // 4,259,840 elems per freq plane
#define RED   (DIM/32)             // 8

typedef __attribute__((ext_vector_type(16))) __bf16    v16bf;
typedef __attribute__((ext_vector_type(8)))  float     v8f;
typedef __attribute__((ext_vector_type(8)))  unsigned  v8u;

// pack two f32 into one dword of bf16 pair: lo -> [15:0], hi -> [31:16]
// (matches ISA 7.12.2: even K in low half of the VGPR, odd K in high half)
__device__ __forceinline__ unsigned pack2bf(float lo, float hi) {
    unsigned a = __builtin_bit_cast(unsigned, lo);
    a += 0x7FFFu + ((a >> 16) & 1u);               // RNE
    unsigned b = __builtin_bit_cast(unsigned, hi);
    b += 0x7FFFu + ((b >> 16) & 1u);
    return (a >> 16) | (b & 0xFFFF0000u);
}

// ---------------------------------------------------------------------------
// WMMA tile helper: stages a 16x32 A tile and 32x16 B tile (f32 -> packed
// bf16 pairs) through a per-wave private LDS slice, gathers the per-lane
// VGPR layouts as packed dwords, and issues v_wmma_f32_16x16x32_bf16.
// A[m][k] = A[m*ars + k*acs],  B[k][n] = B[k*brs + n*bcs].
// sA: 256 dwords (16 rows x 16 k-pairs), sB: 256 dwords (16 k-pairs x 16 n).
// Per-wave private slices => only wave-level ordering needed (DS ops are
// in-order within a wave), so a scheduling barrier suffices.
// ---------------------------------------------------------------------------
__device__ __forceinline__ v8f wmma_mac_tile(
        const float* __restrict__ A, long ars, long acs,
        const float* __restrict__ B, long brs, long bcs,
        v8f c, unsigned* sA, unsigned* sB)
{
    const int lane = threadIdx.x & 31;
    #pragma unroll
    for (int i = 0; i < 8; ++i) {
        int p   = lane * 8 + i;            // 0..255 packed pairs, each once
        int am  = p >> 4, akp = p & 15;    // A: row, k-pair
        sA[p] = pack2bf(A[(long)am * ars + (long)(2 * akp)     * acs],
                        A[(long)am * ars + (long)(2 * akp + 1) * acs]);
        int bkp = p >> 4, bn = p & 15;     // B: k-pair, col
        sB[p] = pack2bf(B[(long)(2 * bkp)     * brs + (long)bn * bcs],
                        B[(long)(2 * bkp + 1) * brs + (long)bn * bcs]);
    }
    __builtin_amdgcn_wave_barrier();

    v8u ua, ub;
    const int g    = lane >> 4;
    const int mrow = lane & 15;
    const int ncol = lane & 15;
    #pragma unroll
    for (int v = 0; v < 8; ++v) {
        // A 16-bit 16x32 layout (ISA 7.12.2), pair index = k/2
        int kap = (v < 4) ? (g * 4 + v) : (8 + g * 4 + (v - 4));
        ua[v] = sA[mrow * 16 + kap];
        // B 32x16: same K layout as A (pair index = k/2)
        int kbp = (v < 4) ? (g * 4 + v) : (8 + g * 4 + (v - 4));
        ub[v] = sB[kbp * 16 + ncol];
    }
    __builtin_amdgcn_wave_barrier();

    c = __builtin_amdgcn_wmma_f32_16x16x32_bf16(
            /*neg_a=*/false, __builtin_bit_cast(v16bf, ua),
            /*neg_b=*/false, __builtin_bit_cast(v16bf, ub),
            /*c_mod=*/(short)0, c, /*reuse_a=*/false, /*reuse_b=*/false);
    return c;
}

// ---------------------------------------------------------------------------
// conv1x1 as GEMM:  Y[b,o,p] = sum_c Wt[o,c] * X[b,c,p] (+bias) (+=Y if acc)
// M=O=256, K=C=256, N=HW per batch.  One wave per 16x16 output tile.
// ---------------------------------------------------------------------------
__global__ __launch_bounds__(256)
void conv1x1_wmma(const float* __restrict__ X,
                  const float* __restrict__ Wt,
                  const float* __restrict__ bias,
                  float* __restrict__ Y, int acc)
{
    __shared__ unsigned sA[8 * 256];
    __shared__ unsigned sB[8 * 256];
    const int wave = threadIdx.x >> 5;
    const int lane = threadIdx.x & 31;
    int t  = blockIdx.x * 8 + wave;              // 0 .. 32767
    int b  = t >> 14;                            // / (16*1024)
    int r  = t & 16383;
    int mt = r >> 10;                            // 0..15
    int nt = r & 1023;                           // 0..1023
    long m0 = (long)mt * 16;
    long n0 = (long)nt * 16;

    v8f c = {};
    #pragma unroll 1
    for (int k0 = 0; k0 < DIM; k0 += 32) {
        if (k0 + 32 < DIM) {
            // gfx1250 global_prefetch for the next K-slab of X
            __builtin_prefetch(&X[((long)b * DIM + k0 + 32) * HW + n0], 0, 1);
        }
        c = wmma_mac_tile(&Wt[m0 * DIM + k0], DIM, 1,
                          &X[((long)b * DIM + k0) * HW + n0], HW, 1,
                          c, sA + wave * 256, sB + wave * 256);
    }
    const int g = lane >> 4, n = lane & 15;
    #pragma unroll
    for (int v = 0; v < 8; ++v) {
        long o   = m0 + v + 8 * g;
        long idx = ((long)b * DIM + o) * HW + n0 + n;
        float val = c[v] + (bias ? bias[o] : 0.0f);
        if (acc) Y[idx] += val; else Y[idx] = val;
    }
}

// ---------------------------------------------------------------------------
// Attention scores: S[bh, c, d] = sum_n Q[row_c, n] * K[row_d, n]
// Q,K: [B*DIM, HW] row-major; rows grouped 32 per (b,h).  64 wave-tiles.
// ---------------------------------------------------------------------------
__global__ __launch_bounds__(256)
void attn_scores_wmma(const float* __restrict__ Q,
                      const float* __restrict__ K,
                      float* __restrict__ S)
{
    __shared__ unsigned sA[8 * 256];
    __shared__ unsigned sB[8 * 256];
    const int wave = threadIdx.x >> 5;
    const int lane = threadIdx.x & 31;
    int t  = blockIdx.x * 8 + wave;             // 0..63
    int bh = t >> 2;
    int mt = (t >> 1) & 1;
    int nt = t & 1;
    int b = bh >> 3, h = bh & 7;
    long rowbase = (long)b * DIM + (long)h * 32;

    v8f c = {};
    #pragma unroll 1
    for (long k0 = 0; k0 < HW; k0 += 32) {
        c = wmma_mac_tile(&Q[(rowbase + mt * 16) * HW + k0], HW, 1,
                          &K[(rowbase + nt * 16) * HW + k0], 1, HW,   // B = K^T
                          c, sA + wave * 256, sB + wave * 256);
    }
    const int g = lane >> 4, n = lane & 15;
    #pragma unroll
    for (int v = 0; v < 8; ++v) {
        int m = mt * 16 + v + 8 * g;
        S[((long)bh * 32 + m) * 32 + nt * 16 + n] = c[v];
    }
}

// ---------------------------------------------------------------------------
// out_g = A1 @ V2 + A2 @ V1   per (b,h):  [32x32]x[32x16384]; two WMMAs into
// one accumulator (K = 32, exactly one 16x16x32 step each).
// ---------------------------------------------------------------------------
__global__ __launch_bounds__(256)
void attn_out_wmma(const float* __restrict__ A1,
                   const float* __restrict__ V2,
                   const float* __restrict__ A2,
                   const float* __restrict__ V1,
                   float* __restrict__ OutG)
{
    __shared__ unsigned sA[8 * 256];
    __shared__ unsigned sB[8 * 256];
    const int wave = threadIdx.x >> 5;
    const int lane = threadIdx.x & 31;
    int t  = blockIdx.x * 8 + wave;          // 0..32767
    int bh = t >> 11;
    int r  = t & 2047;
    int mt = r >> 10;
    int nt = r & 1023;
    int b = bh >> 3, h = bh & 7;
    long rowbase = (long)b * DIM + (long)h * 32;
    long n0 = (long)nt * 16;

    v8f c = {};
    c = wmma_mac_tile(&A1[((long)bh * 32 + mt * 16) * 32], 32, 1,
                      &V2[rowbase * HW + n0], HW, 1,
                      c, sA + wave * 256, sB + wave * 256);
    c = wmma_mac_tile(&A2[((long)bh * 32 + mt * 16) * 32], 32, 1,
                      &V1[rowbase * HW + n0], HW, 1,
                      c, sA + wave * 256, sB + wave * 256);

    const int g = lane >> 4, n = lane & 15;
    #pragma unroll
    for (int v = 0; v < 8; ++v) {
        long row = rowbase + mt * 16 + v + 8 * g;
        OutG[row * HW + n0 + n] = c[v];
    }
}

// ---------------------------------------------------------------------------
// Channel LayerNorm (WithBias): per-pixel mean/var over 256 channels.
// ---------------------------------------------------------------------------
__global__ void layernorm_c_kernel(const float* __restrict__ x,
                                   const float* __restrict__ w,
                                   const float* __restrict__ bb,
                                   float* __restrict__ y)
{
    long p = (long)blockIdx.x * blockDim.x + threadIdx.x;   // 0 .. B*HW-1
    long b = p >> 14;
    long pix = p & (HW - 1);
    const float* xp = x + (b * DIM) * (long)HW + pix;
    float s = 0.f, ss = 0.f;
    for (int c = 0; c < DIM; ++c) {
        float v = xp[(long)c * HW];
        s += v; ss += v * v;
    }
    float mu  = s * (1.0f / DIM);
    float var = ss * (1.0f / DIM) - mu * mu;
    float inv = rsqrtf(var + 1e-5f);
    float* yp = y + (b * DIM) * (long)HW + pix;
    for (int c = 0; c < DIM; ++c)
        yp[(long)c * HW] = (xp[(long)c * HW] - mu) * inv * w[c] + bb[c];
}

// ---------------------------------------------------------------------------
// Depthwise KxK conv, SAME zero padding, optional bias.
// ---------------------------------------------------------------------------
__global__ void dwconv_kernel(const float* __restrict__ in,
                              const float* __restrict__ w,
                              const float* __restrict__ bias,
                              float* __restrict__ out, int K)
{
    long idx = (long)blockIdx.x * 256 + threadIdx.x;   // B*C*H*W
    int x  = (int)(idx & (WW - 1));
    int y  = (int)((idx >> 7) & (HH - 1));
    long bc = idx >> 14;
    int c  = (int)(bc & (DIM - 1));
    const float* ip = in + bc * (long)HW;
    const float* wp = w + (long)c * K * K;
    int pad = K >> 1;
    float acc = bias ? bias[c] : 0.f;
    for (int ky = 0; ky < K; ++ky) {
        int yy = y + ky - pad;
        if ((unsigned)yy >= (unsigned)HH) continue;
        for (int kx = 0; kx < K; ++kx) {
            int xx = x + kx - pad;
            if ((unsigned)xx >= (unsigned)WW) continue;
            acc += ip[yy * WW + xx] * wp[ky * K + kx];
        }
    }
    out[idx] = acc;
}

// ---------------------------------------------------------------------------
// L2-normalize each 16384-length row (in place).  grid = B*DIM rows.
// ---------------------------------------------------------------------------
__global__ void l2norm_rows(float* __restrict__ x)
{
    __shared__ float red[256];
    float* xp = x + (long)blockIdx.x * HW;
    float ss = 0.f;
    for (int i = threadIdx.x; i < HW; i += 256) { float v = xp[i]; ss += v * v; }
    red[threadIdx.x] = ss; __syncthreads();
    for (int s = 128; s > 0; s >>= 1) {
        if ((int)threadIdx.x < s) red[threadIdx.x] += red[threadIdx.x + s];
        __syncthreads();
    }
    float inv = 1.0f / fmaxf(sqrtf(red[0]), 1e-12f);
    for (int i = threadIdx.x; i < HW; i += 256) xp[i] *= inv;
}

// ---------------------------------------------------------------------------
// Row-wise softmax over 32 with temperature; one wave per row.
// ---------------------------------------------------------------------------
__global__ void softmax_rows(const float* __restrict__ S,
                             const float* __restrict__ temp,
                             float* __restrict__ A)
{
    int row  = blockIdx.x;                 // b*8*32 rows
    int h    = (row >> 5) & 7;
    int lane = threadIdx.x;
    float v  = S[(long)row * 32 + lane] * temp[h];
    float m  = v;
    for (int o = 16; o > 0; o >>= 1) m = fmaxf(m, __shfl_xor(m, o, 32));
    float e = __expf(v - m);
    float s = e;
    for (int o = 16; o > 0; o >>= 1) s += __shfl_xor(s, o, 32);
    A[(long)row * 32 + lane] = e / s;
}

// ---------------------------------------------------------------------------
// Elementwise helpers
// ---------------------------------------------------------------------------
__global__ void eadd(const float* __restrict__ a, const float* __restrict__ b,
                     float* __restrict__ o)
{
    long i = (long)blockIdx.x * 256 + threadIdx.x;
    o[i] = a[i] + b[i];
}
__global__ void eadd_acc(float* __restrict__ o, const float* __restrict__ a,
                         const float* __restrict__ b)
{
    long i = (long)blockIdx.x * 256 + threadIdx.x;
    o[i] = o[i] + a[i] + b[i];
}

// ---------------------------------------------------------------------------
// 128-point serial radix-2 FFT on an LDS row (stride 1).
// ---------------------------------------------------------------------------
__device__ void fft128(float2* d, bool inv)
{
    for (int i = 1; i < 127; ++i) {
        int j = (int)(__brev((unsigned)i) >> 25);
        if (j > i) { float2 t = d[i]; d[i] = d[j]; d[j] = t; }
    }
    for (int len = 2; len <= 128; len <<= 1) {
        int half = len >> 1;
        float ang0 = (inv ? 6.28318530717958647692f
                          : -6.28318530717958647692f) / (float)len;
        for (int base = 0; base < 128; base += len) {
            for (int j = 0; j < half; ++j) {
                float sn, cs; __sincosf(ang0 * (float)j, &sn, &cs);
                float2 u = d[base + j];
                float2 t = d[base + j + half];
                float tr = cs * t.x - sn * t.y;
                float ti = cs * t.y + sn * t.x;
                d[base + j]        = make_float2(u.x + tr, u.y + ti);
                d[base + j + half] = make_float2(u.x - tr, u.y - ti);
            }
        }
    }
}

#define ORTHO 0.08838834764831845f   // 1/sqrt(128)

// forward FFT over w for every row; real input -> full complex [.,128]
__global__ void fft_rows_fwd(const float* __restrict__ in,
                             float* __restrict__ mre, float* __restrict__ mim)
{
    __shared__ float2 srow[32][128];
    long r = (long)blockIdx.x * 32 + threadIdx.x;   // B*C*H rows
    float2* d = srow[threadIdx.x];
    const float* src = in + r * WW;
    for (int i = 0; i < 128; ++i) d[i] = make_float2(src[i], 0.f);
    fft128(d, false);
    float* orp = mre + r * WW; float* oip = mim + r * WW;
    for (int i = 0; i < 128; ++i) { orp[i] = d[i].x * ORTHO; oip[i] = d[i].y * ORTHO; }
}

// forward FFT over h for columns 0..64 -> [b,c,128,65]
__global__ void fft_cols_fwd(const float* __restrict__ mre,
                             const float* __restrict__ mim,
                             float* __restrict__ tre, float* __restrict__ tim)
{
    __shared__ float2 srow[32][128];
    long cid = (long)blockIdx.x * 32 + threadIdx.x;   // B*C*65 columns
    if (cid >= (long)BATCH * DIM * FW) return;
    long bc = cid / FW; int col = (int)(cid % FW);
    float2* d = srow[threadIdx.x];
    const float* ir = mre + bc * (long)HW + col;
    const float* ii = mim + bc * (long)HW + col;
    for (int i = 0; i < 128; ++i) d[i] = make_float2(ir[i * WW], ii[i * WW]);
    fft128(d, false);
    for (int i = 0; i < 128; ++i) {
        long o = (bc * HH + i) * FW + col;
        tre[o] = d[i].x * ORTHO; tim[o] = d[i].y * ORTHO;
    }
}

// inverse FFT over h (65 columns) -> intermediate [b,c,128,65]
__global__ void fft_cols_inv(const float* __restrict__ tre,
                             const float* __restrict__ tim,
                             float* __restrict__ mre, float* __restrict__ mim)
{
    __shared__ float2 srow[32][128];
    long cid = (long)blockIdx.x * 32 + threadIdx.x;
    if (cid >= (long)BATCH * DIM * FW) return;
    long bc = cid / FW; int col = (int)(cid % FW);
    float2* d = srow[threadIdx.x];
    for (int i = 0; i < 128; ++i) {
        long o = (bc * HH + i) * FW + col;
        d[i] = make_float2(tre[o], tim[o]);
    }
    fft128(d, true);
    for (int i = 0; i < 128; ++i) {
        long o = (bc * HH + i) * FW + col;
        mre[o] = d[i].x * ORTHO; mim[o] = d[i].y * ORTHO;
    }
}

// inverse rfft over w: Hermitian-extend 65 -> 128, inverse FFT, keep real
__global__ void fft_rows_inv(const float* __restrict__ mre,
                             const float* __restrict__ mim,
                             float* __restrict__ out)
{
    __shared__ float2 srow[32][128];
    long r = (long)blockIdx.x * 32 + threadIdx.x;   // B*C*H rows
    float2* d = srow[threadIdx.x];
    const float* ir = mre + r * FW;
    const float* ii = mim + r * FW;
    for (int w = 0; w < FW; ++w) d[w] = make_float2(ir[w], ii[w]);
    for (int w = FW; w < 128; ++w) d[w] = make_float2(ir[128 - w], -ii[128 - w]);
    fft128(d, true);
    float* op = out + r * WW;
    for (int i = 0; i < 128; ++i) op[i] = d[i].x * ORTHO;
}

// ---------------------------------------------------------------------------
// FFT gate MLP (256 -> 8 -> 256) with training-mode BatchNorm.
// ---------------------------------------------------------------------------
__global__ void gate_proj1(const float* __restrict__ re,
                           const float* __restrict__ w1,
                           const float* __restrict__ b1,
                           float* __restrict__ y)
{
    long i = (long)blockIdx.x * 256 + threadIdx.x;   // B*RED*FPX
    if (i >= (long)BATCH * RED * FPX) return;
    int p = (int)(i % FPX);
    int r = (int)((i / FPX) % RED);
    long b = i / ((long)FPX * RED);
    const float* xp = re + (b * DIM) * (long)FPX + p;
    const float* wr = w1 + (long)r * DIM;
    float acc = b1[r];
    for (int c = 0; c < DIM; ++c) acc += xp[(long)c * FPX] * wr[c];
    y[i] = acc;
}

__global__ void bn_stats(const float* __restrict__ y, float* __restrict__ stats)
{
    __shared__ float s1[256], s2[256];
    int r = blockIdx.x;                      // RED blocks
    float a = 0.f, q = 0.f;
    for (int b = 0; b < BATCH; ++b) {
        const float* yp = y + ((long)b * RED + r) * FPX;
        for (int i = threadIdx.x; i < FPX; i += 256) { float v = yp[i]; a += v; q += v * v; }
    }
    s1[threadIdx.x] = a; s2[threadIdx.x] = q; __syncthreads();
    for (int s = 128; s > 0; s >>= 1) {
        if ((int)threadIdx.x < s) { s1[threadIdx.x] += s1[threadIdx.x + s];
                                    s2[threadIdx.x] += s2[threadIdx.x + s]; }
        __syncthreads();
    }
    if (threadIdx.x == 0) {
        float n  = (float)(BATCH * FPX);
        float mu = s1[0] / n;
        stats[r]       = mu;
        stats[RED + r] = s2[0] / n - mu * mu;
    }
}

__global__ void bn_relu(float* __restrict__ y, const float* __restrict__ stats,
                        const float* __restrict__ g, const float* __restrict__ beta)
{
    long i = (long)blockIdx.x * 256 + threadIdx.x;
    if (i >= (long)BATCH * RED * FPX) return;
    int r = (int)((i / FPX) % RED);
    float v = (y[i] - stats[r]) * rsqrtf(stats[RED + r] + 1e-5f) * g[r] + beta[r];
    y[i] = fmaxf(v, 0.f);
}

__global__ void gate_proj2(const float* __restrict__ y,
                           const float* __restrict__ w2,
                           const float* __restrict__ b2,
                           float* __restrict__ gate, int acc)
{
    long i = (long)blockIdx.x * 256 + threadIdx.x;   // B*DIM*FPX
    if (i >= (long)BATCH * DIM * FPX) return;
    int p = (int)(i % FPX);
    int c = (int)((i / FPX) % DIM);
    long b = i / ((long)FPX * DIM);
    const float* yp = y + (b * RED) * (long)FPX + p;
    const float* wc = w2 + (long)c * RED;
    float a = b2[c];
    #pragma unroll
    for (int r = 0; r < RED; ++r) a += yp[(long)r * FPX] * wc[r];
    float s = 1.0f / (1.0f + __expf(-a));
    if (acc) gate[i] += s; else gate[i] = s;
}

// xf = (1 + gate) * (tf + cf)   (gate is real; applied to re and im)
__global__ void xf_combine(float* __restrict__ tre, float* __restrict__ tim,
                           const float* __restrict__ cre, const float* __restrict__ cim,
                           const float* __restrict__ gate)
{
    long i = (long)blockIdx.x * 256 + threadIdx.x;
    if (i >= (long)FE) return;
    float g1 = 1.0f + gate[i];
    tre[i] = g1 * (tre[i] + cre[i]);
    tim[i] = g1 * (tim[i] + cim[i]);
}

// ---------------------------------------------------------------------------
// Host-side orchestration
// ---------------------------------------------------------------------------
extern "C" void kernel_launch(void* const* d_in, const int* in_sizes, int n_in,
                              void* d_out, int out_size, void* d_ws, size_t ws_size,
                              hipStream_t stream)
{
    const float* x1     = (const float*)d_in[0];
    const float* x2     = (const float*)d_in[1];
    const float* ln_w   = (const float*)d_in[2];
    const float* ln_b   = (const float*)d_in[3];
    const float* out_pw = (const float*)d_in[4];
    const float* temp   = (const float*)d_in[5];
    const float* qkv_w  = (const float*)d_in[6];
    const float* qkv_b  = (const float*)d_in[7];
    const float* dwq1   = (const float*)d_in[8];
    const float* dwk1   = (const float*)d_in[9];
    const float* dwv1   = (const float*)d_in[10];
    const float* dwq2   = (const float*)d_in[11];
    const float* dwk2   = (const float*)d_in[12];
    const float* dwv2   = (const float*)d_in[13];
    const float* attn_w = (const float*)d_in[14];
    const float* lpw    = (const float*)d_in[15];
    const float* lpb    = (const float*)d_in[16];
    const float* ld3w   = (const float*)d_in[17];
    const float* ld3b   = (const float*)d_in[18];
    const float* ld5w   = (const float*)d_in[19];
    const float* ld5b   = (const float*)d_in[20];
    const float* ld7w   = (const float*)d_in[21];
    const float* ld7b   = (const float*)d_in[22];
    const float* low    = (const float*)d_in[23];
    const float* lob    = (const float*)d_in[24];
    const float* fw1    = (const float*)d_in[25];
    const float* fb1    = (const float*)d_in[26];
    const float* fg     = (const float*)d_in[27];
    const float* fbeta  = (const float*)d_in[28];
    const float* fw2    = (const float*)d_in[29];
    const float* fb2    = (const float*)d_in[30];
    float* out = (float*)d_out;

    float* ws = (float*)d_ws;
    float* buf[10];
    for (int i = 0; i < 10; ++i) buf[i] = ws + (size_t)i * NE;
    float* S1    = ws + 10 * NE;
    float* A1    = S1 + 16384;
    float* S2    = A1 + 16384;
    float* A2    = S2 + 16384;
    float* stats = A2 + 16384;                 // 16 floats

    const int GE   = (int)(NE / 256);          // 32768 elementwise blocks
    const int GF   = (int)((FE + 255) / 256);  // 16640 freq elementwise blocks
    const int GEMM = (int)(NE / (16 * 16 * 8));// 4096 blocks (8 waves each)
    const int GROW = (int)(NE / WW / 32);      // 2048 row-FFT blocks
    const int GCOL = (int)((BATCH * DIM * FW + 31) / 32);  // 1040
    const int GG1  = (int)(((long)BATCH * RED * FPX + 255) / 256); // 520

    // ---- LayerNorm ----
    layernorm_c_kernel<<<128, 256, 0, stream>>>(x1, ln_w, ln_b, buf[0]); // n1
    layernorm_c_kernel<<<128, 256, 0, stream>>>(x2, ln_w, ln_b, buf[1]); // n2

    // ---- shared 1x1 qkv projection (WMMA) ----
    conv1x1_wmma<<<GEMM, 256, 0, stream>>>(buf[0], qkv_w, qkv_b, buf[2], 0); // p1
    conv1x1_wmma<<<GEMM, 256, 0, stream>>>(buf[1], qkv_w, qkv_b, buf[3], 0); // p2

    // ---- depthwise q/k/v ----
    dwconv_kernel<<<GE, 256, 0, stream>>>(buf[2], dwq1, nullptr, buf[4], 3); // q1
    dwconv_kernel<<<GE, 256, 0, stream>>>(buf[2], dwk1, nullptr, buf[5], 3); // k1
    dwconv_kernel<<<GE, 256, 0, stream>>>(buf[2], dwv1, nullptr, buf[6], 3); // v1
    dwconv_kernel<<<GE, 256, 0, stream>>>(buf[3], dwq2, nullptr, buf[7], 3); // q2
    dwconv_kernel<<<GE, 256, 0, stream>>>(buf[3], dwk2, nullptr, buf[8], 3); // k2
    dwconv_kernel<<<GE, 256, 0, stream>>>(buf[3], dwv2, nullptr, buf[9], 3); // v2

    // ---- L2 norms ----
    l2norm_rows<<<BATCH * DIM, 256, 0, stream>>>(buf[4]);
    l2norm_rows<<<BATCH * DIM, 256, 0, stream>>>(buf[5]);
    l2norm_rows<<<BATCH * DIM, 256, 0, stream>>>(buf[7]);
    l2norm_rows<<<BATCH * DIM, 256, 0, stream>>>(buf[8]);

    // ---- attention scores + softmax (WMMA, K=16384) ----
    attn_scores_wmma<<<8, 256, 0, stream>>>(buf[4], buf[5], S1);
    attn_scores_wmma<<<8, 256, 0, stream>>>(buf[7], buf[8], S2);
    softmax_rows<<<BATCH * HEADS * 32, 32, 0, stream>>>(S1, temp, A1);
    softmax_rows<<<BATCH * HEADS * 32, 32, 0, stream>>>(S2, temp, A2);

    // ---- out_g = A1@v2 + A2@v1 (fused, two WMMAs per tile) ----
    attn_out_wmma<<<GEMM, 256, 0, stream>>>(A1, buf[9], A2, buf[6], buf[2]);

    // ---- local block on n1 then n2, accumulated into buf[8] ----
    for (int pass = 0; pass < 2; ++pass) {
        const float* nsrc = pass ? buf[1] : buf[0];
        conv1x1_wmma<<<GEMM, 256, 0, stream>>>(nsrc, lpw, lpb, buf[3], 0);   // a
        dwconv_kernel<<<GE, 256, 0, stream>>>(buf[3], ld3w, ld3b, buf[4], 3);// x3
        eadd<<<GE, 256, 0, stream>>>(buf[3], buf[4], buf[5]);                // a+x3
        dwconv_kernel<<<GE, 256, 0, stream>>>(buf[5], ld5w, ld5b, buf[6], 5);// x5
        eadd<<<GE, 256, 0, stream>>>(buf[3], buf[6], buf[5]);                // a+x5
        dwconv_kernel<<<GE, 256, 0, stream>>>(buf[5], ld7w, ld7b, buf[7], 7);// x7
        eadd_acc<<<GE, 256, 0, stream>>>(buf[4], buf[6], buf[7]);            // x3+x5+x7
        conv1x1_wmma<<<GEMM, 256, 0, stream>>>(buf[4], low, lob, buf[8], pass);
    }

    // ---- forward FFTs: out_g(buf2)->tf(buf4,buf5), out_l(buf8)->cf(buf6,buf7)
    fft_rows_fwd<<<GROW, 32, 0, stream>>>(buf[2], buf[0], buf[1]);
    fft_cols_fwd<<<GCOL, 32, 0, stream>>>(buf[0], buf[1], buf[4], buf[5]);
    fft_rows_fwd<<<GROW, 32, 0, stream>>>(buf[8], buf[0], buf[1]);
    fft_cols_fwd<<<GCOL, 32, 0, stream>>>(buf[0], buf[1], buf[6], buf[7]);

    // ---- gate = g(tf.re) + g(cf.re) ----  (ygate in buf[3], gate in buf[9])
    gate_proj1<<<GG1, 256, 0, stream>>>(buf[4], fw1, fb1, buf[3]);
    bn_stats<<<RED, 256, 0, stream>>>(buf[3], stats);
    bn_relu<<<GG1, 256, 0, stream>>>(buf[3], stats, fg, fbeta);
    gate_proj2<<<GF, 256, 0, stream>>>(buf[3], fw2, fb2, buf[9], 0);
    gate_proj1<<<GG1, 256, 0, stream>>>(buf[6], fw1, fb1, buf[3]);
    bn_stats<<<RED, 256, 0, stream>>>(buf[3], stats);
    bn_relu<<<GG1, 256, 0, stream>>>(buf[3], stats, fg, fbeta);
    gate_proj2<<<GF, 256, 0, stream>>>(buf[3], fw2, fb2, buf[9], 1);

    // ---- xf = (1+gate)*(tf+cf), then irfft2 -> x_fused (buf2) ----
    xf_combine<<<GF, 256, 0, stream>>>(buf[4], buf[5], buf[6], buf[7], buf[9]);
    fft_cols_inv<<<GCOL, 32, 0, stream>>>(buf[4], buf[5], buf[0], buf[1]);
    fft_rows_inv<<<GROW, 32, 0, stream>>>(buf[0], buf[1], buf[2]);

    // ---- final projections + residual ----
    conv1x1_wmma<<<GEMM, 256, 0, stream>>>(buf[2], attn_w, nullptr, buf[3], 0);
    conv1x1_wmma<<<GEMM, 256, 0, stream>>>(buf[3], out_pw, nullptr, out, 0);
    eadd_acc<<<GE, 256, 0, stream>>>(out, x1, x2);
}
